// HashMemory_39659728011625
// MI455X (gfx1250) — matrix operand order repaired and checked by reading, not compiled, not run on hardware
//
#include <hip/hip_runtime.h>
#include <hip/hip_bf16.h>

// ---- problem constants -----------------------------------------------------
#define BB   8
#define SS   4096
#define ED   1024      // EMBED_DIM
#define DM   512       // MEMORY_DIM
#define WIN  64        // NUM_SLOTS == sliding window
constexpr float SCALE = 0.04419417382415922f;   // 512^-0.5

// ---- CDNA5 WMMA types ------------------------------------------------------
typedef __attribute__((ext_vector_type(16))) __bf16 v16bf;
typedef __attribute__((ext_vector_type(8)))  __bf16 v8bf;
typedef __attribute__((ext_vector_type(8)))  float  v8f;

__device__ inline v8f wmma_bf16(v16bf a, v16bf b, v8f c) {
  // D = A(16x32 bf16) * B(32x16 bf16) + C(16x16 f32)
  return __builtin_amdgcn_wmma_f32_16x16x32_bf16(
      /*neg_a*/false, a, /*neg_b*/false, b,
      /*c_mod*/(short)0, c, /*reuse_a*/false, /*reuse_b*/false);
}

// A fragment: lane holds row M=lane&15; 16 consecutive K elems {base..+15}
// (caller passes base already including the half-wave +16 shift)
__device__ inline v16bf load_a_frag(const __bf16* p) {
  v8bf lo = *reinterpret_cast<const v8bf*>(p);
  v8bf hi = *reinterpret_cast<const v8bf*>(p + 8);
  v16bf r;
#pragma unroll
  for (int i = 0; i < 8; ++i) { r[i] = lo[i]; r[8 + i] = hi[i]; }
  return r;
}

// B fragment: lane holds col N=lane&15; 16 consecutive K values (contiguous)
__device__ inline v16bf load_b_frag(const __bf16* p) {
  return *reinterpret_cast<const v16bf*>(p);
}

__device__ inline unsigned int bf_bits(float f) {
  return (unsigned int)__builtin_bit_cast(unsigned short, (__bf16)f);
}
__device__ inline unsigned int pack2(float lo, float hi) {
  return bf_bits(lo) | (bf_bits(hi) << 16);
}

// ---- kernel 1: transpose + convert weights to bf16 -------------------------
__global__ void prep_weights(const float* __restrict__ Wq,
                             const float* __restrict__ Ww,
                             const float* __restrict__ Wo,
                             __bf16* __restrict__ wqt,   // [DM][ED]
                             __bf16* __restrict__ wwt,   // [DM][ED]
                             __bf16* __restrict__ wot) { // [ED][DM]
  int idx = blockIdx.x * blockDim.x + threadIdx.x;
  if (idx >= ED * DM) return;
  int n = idx / ED, e = idx % ED;          // Wt[n][e] = W[e][n]
  wqt[idx] = (__bf16)Wq[e * DM + n];
  wwt[idx] = (__bf16)Ww[e * DM + n];
  int eo = idx / DM, ko = idx % DM;        // Wot[e][k] = Wo[k][e]
  wot[idx] = (__bf16)Wo[ko * ED + eo];
}

// ---- kernel 2: fused Q / WV projection GEMM --------------------------------
// grid (2048 row-tiles, 2), block 128.
// Block stages one 16x1024 bf16 A tile in LDS (converted once); each wave
// computes 4 N-tiles (64 cols) for BOTH Q and WV -> 8 WMMAs per K-chunk.
__global__ void __launch_bounds__(128)
proj_kernel(const float* __restrict__ emb,
            const float* __restrict__ bq, const float* __restrict__ bw,
            const __bf16* __restrict__ wqt, const __bf16* __restrict__ wwt,
            __bf16* __restrict__ Q,    // [B][S][DM]
            __bf16* __restrict__ WV,   // [B][S][DM]
            __bf16* __restrict__ WVt)  // [B][DM][S]
{
  __shared__ __align__(16) __bf16 smem[16 * ED];   // 32 KB
  const int t    = threadIdx.x;
  const int lane = t & 31;
  const int w    = t >> 5;
  const int rt   = blockIdx.x;
  const int b    = rt >> 8;
  const int s0   = (rt & 255) * 16;
  const int slotbase = blockIdx.y * 16 + w * 4;    // first N-tile of this wave
  const int ncol = lane & 15;
  const int half = lane >> 4;

  // ---- cooperative stage: emb fp32 -> bf16 into LDS (once per block) ----
  {
    int row = t >> 3, seg = t & 7;                 // 128 floats per thread
    const float4* src = reinterpret_cast<const float4*>(
        emb + ((size_t)(b * SS + s0 + row)) * ED + seg * 128);
    unsigned int* dst =
        reinterpret_cast<unsigned int*>(&smem[row * ED + seg * 128]);
#pragma unroll
    for (int i = 0; i < 32; ++i) {
      float4 f = src[i];
      dst[2 * i]     = pack2(f.x, f.y);
      dst[2 * i + 1] = pack2(f.z, f.w);
    }
  }
  __syncthreads();

  // hoisted, loop-invariant B base pointers (K offset folds into immediates)
  const __bf16* bqbase[4];
  const __bf16* bwbase[4];
#pragma unroll
  for (int j = 0; j < 4; ++j) {
    size_t bo = (size_t)((slotbase + j) * 16 + ncol) * ED + half * 16;
    bqbase[j] = wqt + bo;
    bwbase[j] = wwt + bo;
  }

  v8f cq[4] = {}; v8f cw[4] = {};
  const __bf16* arow = &smem[ncol * ED + half * 16];

#pragma unroll 8
  for (int k0 = 0; k0 < ED; k0 += 32) {
    v16bf a = load_a_frag(arow + k0);
#pragma unroll
    for (int j = 0; j < 4; ++j) {
      cq[j] = wmma_bf16(a, load_b_frag(bqbase[j] + k0), cq[j]);
      cw[j] = wmma_bf16(a, load_b_frag(bwbase[j] + k0), cw[j]);
    }
  }

  float biasq[4], biasw[4];
#pragma unroll
  for (int j = 0; j < 4; ++j) {
    int col = (slotbase + j) * 16 + ncol;
    biasq[j] = bq[col];
    biasw[j] = bw[col];
  }

  // ---- WVt packed b128 stores straight from registers ----
#pragma unroll
  for (int j = 0; j < 4; ++j) {
    unsigned int pk0 = pack2(cw[j][0] + biasw[j], cw[j][1] + biasw[j]);
    unsigned int pk1 = pack2(cw[j][2] + biasw[j], cw[j][3] + biasw[j]);
    unsigned int pk2 = pack2(cw[j][4] + biasw[j], cw[j][5] + biasw[j]);
    unsigned int pk3 = pack2(cw[j][6] + biasw[j], cw[j][7] + biasw[j]);
    size_t toff =
        ((size_t)(b * DM + (slotbase + j) * 16 + ncol)) * SS + s0 + half * 8;
    *reinterpret_cast<uint4*>(WVt + toff) = make_uint4(pk0, pk1, pk2, pk3);
  }

  // ---- transpose C tiles through LDS, then coalesced b128 stores ----
  __syncthreads();                                  // done reading shared A
  __bf16* qs = &smem[w * 2048];                     // [16][64] per wave
  __bf16* ws = &smem[w * 2048 + 1024];              // [16][64] per wave
#pragma unroll
  for (int j = 0; j < 4; ++j)
#pragma unroll
    for (int v = 0; v < 8; ++v) {
      int m = v + half * 8;
      qs[m * 64 + j * 16 + ncol] = (__bf16)(cq[j][v] + biasq[j]);
      ws[m * 64 + j * 16 + ncol] = (__bf16)(cw[j][v] + biasw[j]);
    }
  {
    int row = lane >> 1, part = lane & 1;           // 2 lanes per row
    size_t goff = ((size_t)(b * SS + s0 + row)) * DM + slotbase * 16 + part * 32;
    const uint4* q4 = reinterpret_cast<const uint4*>(&qs[row * 64 + part * 32]);
    const uint4* w4 = reinterpret_cast<const uint4*>(&ws[row * 64 + part * 32]);
    uint4* gq = reinterpret_cast<uint4*>(Q + goff);
    uint4* gw = reinterpret_cast<uint4*>(WV + goff);
#pragma unroll
    for (int i = 0; i < 4; ++i) { gq[i] = q4[i]; gw[i] = w4[i]; }
  }
}

// ---- kernel 3: sliding-window attention (window = previous 64 tokens) ------
// grid 512, block 128: wave w owns query tile = bx*4+w (16 queries)
__global__ void __launch_bounds__(128)
attn_kernel(const __bf16* __restrict__ Q,
            const __bf16* __restrict__ WV,
            const __bf16* __restrict__ WVt,
            __bf16* __restrict__ R)   // retrieved [B][S][DM]
{
  __shared__ float pl[4][16][96];                  // probabilities per wave
  __shared__ __align__(16) __bf16 rbuf[4][16][16]; // store-coalescing tile
  const int lane = threadIdx.x & 31;
  const int w    = threadIdx.x >> 5;
  const int tile = blockIdx.x * 4 + w;
  const int b    = tile >> 8;
  const int t0   = tile & 255;
  const int s0   = t0 * 16;
  const int ncol = lane & 15;
  const int half = lane >> 4;
  const int KB   = s0 - 64;                        // first key of 5-tile span

  // ---- scores: 5 key tiles x (DM/32) WMMAs, Q fragment reused ----
  const __bf16* kbase[5];
#pragma unroll
  for (int i = 0; i < 5; ++i)
    kbase[i] = WV + ((size_t)(b * SS + (t0 - 4 + i) * 16 + ncol)) * DM + half * 16;

  v8f acc[5] = {};
  const __bf16* qrow = Q + ((size_t)(b * SS + s0 + ncol)) * DM + half * 16;
#pragma unroll 4
  for (int k0 = 0; k0 < DM; k0 += 32) {
    v16bf a = load_a_frag(qrow + k0);
#pragma unroll
    for (int i = 0; i < 5; ++i) {
      if (t0 - 4 + i < 0) continue;                // wave-uniform branch
      acc[i] = wmma_bf16(a, load_b_frag(kbase[i] + k0), acc[i]);
    }
  }

  // ---- mask + softmax over 80 key columns ----
  const float NEG = -3.0e38f;
  float ev[5][8], m[8], sum[8], inv[8];
#pragma unroll
  for (int v = 0; v < 8; ++v) m[v] = NEG;
#pragma unroll
  for (int i = 0; i < 5; ++i) {
    int kt = t0 - 4 + i;
    int j  = kt * 16 + ncol;                       // key index
#pragma unroll
    for (int v = 0; v < 8; ++v) {
      int s = s0 + v + half * 8;                   // query index
      bool valid = (kt >= 0) && (j < s) && (j >= s - WIN);
      float val = valid ? acc[i][v] * SCALE : NEG;
      ev[i][v] = val;
      m[v] = fmaxf(m[v], val);
    }
  }
#pragma unroll
  for (int v = 0; v < 8; ++v) {
#pragma unroll
    for (int off = 1; off < 16; off <<= 1)
      m[v] = fmaxf(m[v], __shfl_xor(m[v], off, 16));
    sum[v] = 0.f;
  }
#pragma unroll
  for (int i = 0; i < 5; ++i)
#pragma unroll
    for (int v = 0; v < 8; ++v) {
      float e = (m[v] <= NEG) ? 0.f : __expf(ev[i][v] - m[v]);
      ev[i][v] = e;
      sum[v] += e;
    }
#pragma unroll
  for (int v = 0; v < 8; ++v) {
#pragma unroll
    for (int off = 1; off < 16; off <<= 1)
      sum[v] += __shfl_xor(sum[v], off, 16);
    inv[v] = (sum[v] > 0.f) ? 1.f / sum[v] : 0.f;  // all-masked row -> zeros
  }

  // ---- stage P through LDS to re-layout C->A ----
#pragma unroll
  for (int v = 0; v < 8; ++v)
    pl[w][v + half * 8][80 + ncol] = 0.f;          // zero-pad keys 80..95
#pragma unroll
  for (int i = 0; i < 5; ++i)
#pragma unroll
    for (int v = 0; v < 8; ++v)
      pl[w][v + half * 8][i * 16 + ncol] = ev[i][v] * inv[v];
  __syncthreads();

  v16bf pa[3];
#pragma unroll
  for (int c = 0; c < 3; ++c)
#pragma unroll
    for (int e = 0; e < 8; ++e) {
      pa[c][e]     = (__bf16)pl[w][ncol][c * 32 + half * 16 + e];
      pa[c][8 + e] = (__bf16)pl[w][ncol][c * 32 + half * 16 + 8 + e];
    }

  // ---- retrieved = P @ V : 32 dim tiles x 3 key chunks ----
  const __bf16* vbase[3];
#pragma unroll
  for (int c = 0; c < 3; ++c) {
    int tok = KB + c * 32 + half * 16;             // clamp: P==0 there anyway
    tok = tok < 0 ? 0 : (tok > SS - 16 ? SS - 16 : tok);
    vbase[c] = WVt + ((size_t)(b * DM + ncol)) * SS + tok;
  }
  const int srow  = lane >> 1, spart = lane & 1;   // store-phase mapping
  __bf16* rout = R + ((size_t)(b * SS + s0 + srow)) * DM + spart * 8;

#pragma unroll 4
  for (int nb = 0; nb < 32; ++nb) {
    v8f r = {};
#pragma unroll
    for (int c = 0; c < 3; ++c)
      r = wmma_bf16(pa[c], load_b_frag(vbase[c] + (size_t)nb * 16 * SS), r);
    // transpose 16x16 tile through LDS -> one b128 store per lane
#pragma unroll
    for (int v = 0; v < 8; ++v)
      rbuf[w][v + half * 8][ncol] = (__bf16)r[v];
    const uint4 rv = *reinterpret_cast<const uint4*>(&rbuf[w][srow][spart * 8]);
    *reinterpret_cast<uint4*>(rout + nb * 16) = rv;
  }
}

// ---- kernel 4: output GEMM (retrieved @ W_out + b_out), fp32 out -----------
// grid (2048, 4), block 128. Block stages 16x512 bf16 A tile in LDS; each
// wave computes 4 N-tiles (64 cols).
__global__ void __launch_bounds__(128)
out_kernel(const __bf16* __restrict__ R, const __bf16* __restrict__ wot,
           const float* __restrict__ bo, float* __restrict__ out) {
  __shared__ __align__(16) __bf16 sm[16 * DM];     // 16 KB (reused for stores)
  const int t    = threadIdx.x;
  const int lane = t & 31;
  const int w    = t >> 5;
  const int rt   = blockIdx.x;
  const int b    = rt >> 8;
  const int s0   = (rt & 255) * 16;
  const int slotbase = blockIdx.y * 16 + w * 4;    // [0,64) N-tiles
  const int ncol = lane & 15;
  const int half = lane >> 4;

  // ---- cooperative stage of A (bf16 copy) ----
  {
    int row = t >> 3, seg = t & 7;                 // 64 elems per thread
    const uint4* src = reinterpret_cast<const uint4*>(
        R + ((size_t)(b * SS + s0 + row)) * DM + seg * 64);
    uint4* dst = reinterpret_cast<uint4*>(&sm[row * DM + seg * 64]);
#pragma unroll
    for (int i = 0; i < 8; ++i) dst[i] = src[i];
  }
  __syncthreads();

  const __bf16* bbase[4];
#pragma unroll
  for (int j = 0; j < 4; ++j)
    bbase[j] = wot + (size_t)((slotbase + j) * 16 + ncol) * DM + half * 16;

  v8f c[4] = {};
  const __bf16* arow = &sm[ncol * DM + half * 16];
#pragma unroll 8
  for (int k0 = 0; k0 < DM; k0 += 32) {
    v16bf a = load_a_frag(arow + k0);
#pragma unroll
    for (int j = 0; j < 4; ++j)
      c[j] = wmma_bf16(a, load_b_frag(bbase[j] + k0), c[j]);
  }

  // ---- transpose through LDS, coalesced float4 stores ----
  __syncthreads();                                 // done reading shared A
  float* osl = reinterpret_cast<float*>(sm) + w * 1024;  // [16][64] f32
#pragma unroll
  for (int j = 0; j < 4; ++j) {
    float bias = bo[(slotbase + j) * 16 + ncol];
#pragma unroll
    for (int v = 0; v < 8; ++v)
      osl[(v + half * 8) * 64 + j * 16 + ncol] = c[j][v] + bias;
  }
  {
    int row = lane >> 1, part = lane & 1;          // 2 lanes per row
    const float4* s4 =
        reinterpret_cast<const float4*>(&osl[row * 64 + part * 32]);
    float4* g4 = reinterpret_cast<float4*>(
        out + ((size_t)(b * SS + s0 + row)) * ED + slotbase * 16 + part * 32);
#pragma unroll
    for (int i = 0; i < 8; ++i) g4[i] = s4[i];
  }
}

// ---- host launcher ---------------------------------------------------------
extern "C" void kernel_launch(void* const* d_in, const int* in_sizes, int n_in,
                              void* d_out, int out_size, void* d_ws, size_t ws_size,
                              hipStream_t stream) {
  (void)in_sizes; (void)n_in; (void)out_size; (void)ws_size;
  const float* emb = (const float*)d_in[0];
  const float* Ww  = (const float*)d_in[1];
  const float* bw  = (const float*)d_in[2];
  const float* Wq  = (const float*)d_in[3];
  const float* bq  = (const float*)d_in[4];
  const float* Wo  = (const float*)d_in[5];
  const float* bo  = (const float*)d_in[6];
  float* out = (float*)d_out;

  char* ws = (char*)d_ws;
  const size_t wBytes = (size_t)ED * DM * 2;       // 1 MB
  const size_t tBytes = (size_t)BB * SS * DM * 2;  // 32 MB
  __bf16* wqt = (__bf16*)(ws);
  __bf16* wwt = (__bf16*)(ws + wBytes);
  __bf16* wot = (__bf16*)(ws + 2 * wBytes);
  __bf16* Qp  = (__bf16*)(ws + 3 * wBytes);
  __bf16* WVp = (__bf16*)(ws + 3 * wBytes + tBytes);
  __bf16* WVt = (__bf16*)(ws + 3 * wBytes + 2 * tBytes);
  __bf16* Rp  = (__bf16*)(ws + 3 * wBytes + 3 * tBytes);

  prep_weights<<<(ED * DM + 255) / 256, 256, 0, stream>>>(Wq, Ww, Wo, wqt, wwt, wot);
  proj_kernel<<<dim3(BB * SS / 16, 2), 128, 0, stream>>>(emb, bq, bw, wqt, wwt,
                                                         Qp, WVp, WVt);
  attn_kernel<<<BB * SS / 16 / 4, 128, 0, stream>>>(Qp, WVp, WVt, Rp);
  out_kernel<<<dim3(BB * SS / 16, 4), 128, 0, stream>>>(Rp, wot, bo, out);
}
